// RNNModel_46368466928025
// MI455X (gfx1250) — matrix operand order repaired and checked by reading, not compiled, not run on hardware
//
#include <hip/hip_runtime.h>
#include <hip/hip_bf16.h>
#include <math.h>

// CTRNN forward for MI455X (gfx1250, wave32, WMMA).
//   x_{t+1} = x_t + a*(-x_t + u_t W_rec^T + I_t W_in^T + b),  u = tanh(x),  a = DT/TAU = 0.2
// Persistent 8-workgroup kernel; W_rec held in VGPRs as bf16 WMMA B-operands
// (one N-tile per wave -> 128 VGPRs, stays under the 256-VGPR direct window);
// per-step u exchange through an L2-resident double buffer + device barrier.

typedef __attribute__((ext_vector_type(16))) __bf16        v16bf;
typedef __attribute__((ext_vector_type(8)))  float         v8f;
typedef __attribute__((ext_vector_type(8)))  unsigned int  v8u;

#define RNN_M   64
#define RNN_N   512
#define NWG     8          // persistent workgroups; each owns 64 output neurons
#define ALPHA   0.2f       // DT / TAU

union BCast { v8u u; v16bf b; };

__device__ __forceinline__ unsigned bf16rne(float f) {
  unsigned u = __float_as_uint(f);
  return (u + 0x7FFFu + ((u >> 16) & 1u)) >> 16;     // round-to-nearest-even to bf16
}
__device__ __forceinline__ unsigned pack2(float lo, float hi) {
  return bf16rne(lo) | (bf16rne(hi) << 16);
}

// A-operand fragment (16x32 bf16): per lane two contiguous 16B chunks of one row.
__device__ __forceinline__ v16bf load_a(const unsigned short* p) {
  const uint4 q0 = *(const uint4*)(p);        // K = kbase   .. kbase+7
  const uint4 q1 = *(const uint4*)(p + 16);   // K = kbase+16 .. kbase+23
  BCast a;
  a.u[0] = q0.x; a.u[1] = q0.y; a.u[2] = q0.z; a.u[3] = q0.w;
  a.u[4] = q1.x; a.u[5] = q1.y; a.u[6] = q1.z; a.u[7] = q1.w;
  return a.b;
}

// Device-wide barrier: monotonic epoch, agent-scope atomics, s_sleep spin.
__device__ __forceinline__ void grid_barrier(unsigned* cnt, unsigned* epoch, unsigned target) {
  __threadfence();                       // release this wave's stores to device scope
  __syncthreads();
  if (threadIdx.x == 0) {
    unsigned prev = __hip_atomic_fetch_add(cnt, 1u, __ATOMIC_ACQ_REL, __HIP_MEMORY_SCOPE_AGENT);
    if (prev == NWG - 1) {
      __hip_atomic_store(cnt, 0u, __ATOMIC_RELAXED, __HIP_MEMORY_SCOPE_AGENT);
      __hip_atomic_store(epoch, target, __ATOMIC_RELEASE, __HIP_MEMORY_SCOPE_AGENT);
    } else {
      while (__hip_atomic_load(epoch, __ATOMIC_ACQUIRE, __HIP_MEMORY_SCOPE_AGENT) < target)
        __builtin_amdgcn_s_sleep(2);
    }
  }
  __syncthreads();
  __threadfence();                       // acquire: invalidate stale near caches
}

__global__ void __launch_bounds__(256, 1)
ctrnn_persistent(const float* __restrict__ I, const float* __restrict__ x0,
                 const float* __restrict__ W_in, const float* __restrict__ W_rec,
                 const float* __restrict__ b,
                 float* __restrict__ x_out, float* __restrict__ u_out,
                 unsigned short* __restrict__ u_ex,   // 2 x (64*512) bf16 exchange buffers
                 float* __restrict__ u_fin,           // final u(T), 64*512 f32
                 unsigned* bar_cnt, unsigned* bar_epoch, int T)
{
  __shared__ float Ish[RNN_M * 2];       // I[:, :, t] staged per step (flat idx = 2*m + c)

  const int tid   = threadIdx.x;
  const int lane  = tid & 31;
  const int wave  = tid >> 5;            // 8 waves
  const int l16   = lane & 15;
  const int hi    = lane >> 4;           // lane half (WMMA layout split)
  const int ntile = wave & 3;            // 4 N-tiles of 16 cover this WG's 64 neurons
  const int mpair = wave >> 2;           // M-tile pair: {2*mpair, 2*mpair+1} cover M=64
  const int g     = blockIdx.x;
  const size_t NT = (size_t)RNN_N * (size_t)T;

  // ---- Per-lane constants and W_rec as bf16 WMMA B operands (one N-tile) ----
  // B tile (K=32 x N=16): lane holds column n = l16, K = kt*32 + (hi?8:0) .. +7 and +16 .. +23,
  // i.e. two 8-element chunks of W_rec row n (since B[k][n] = W_rec[n][k]).
  const int ngl = g * 64 + ntile * 16 + l16;
  v16bf Breg[16];
  {
    const float* wr = W_rec + (size_t)ngl * RNN_N;
    #pragma unroll
    for (int kt = 0; kt < 16; ++kt) {
      const float* wk = wr + kt * 32 + hi * 8;
      BCast cvt;
      #pragma unroll
      for (int e = 0; e < 8; ++e) cvt.u[e] = pack2(wk[2 * e + (e >= 4 ? 8 : 0)], wk[2 * e + 1 + (e >= 4 ? 8 : 0)]);
      Breg[kt] = cvt.b;
    }
  }
  const float win0 = W_in[ngl * 2 + 0];
  const float win1 = W_in[ngl * 2 + 1];
  const float bbn  = b[ngl];

  size_t obase[2];
  #pragma unroll
  for (int a = 0; a < 2; ++a)
    obase[a] = (size_t)((2 * mpair + a) * 16 + hi * 8) * NT + (size_t)ngl * (size_t)T;

  // ---- Initial state: x0 everywhere; emit t=0 outputs; publish u(0) bf16 ----
  const float x0v = x0[0];
  const float u0v = tanhf(x0v);
  float xr[2][8];
  {
    const unsigned short u0b = (unsigned short)bf16rne(u0v);
    #pragma unroll
    for (int a = 0; a < 2; ++a)
      #pragma unroll
      for (int j = 0; j < 8; ++j) {
        xr[a][j] = x0v;
        const int m = (2 * mpair + a) * 16 + hi * 8 + j;
        const size_t o = obase[a] + (size_t)j * NT;      // t = 0
        x_out[o] = x0v;
        u_out[o] = u0v;
        u_ex[(size_t)m * RNN_N + ngl] = u0b;             // buffer 0
      }
  }
  grid_barrier(bar_cnt, bar_epoch, 1u);

  // ---- Time loop ----
  for (int t = 0; t < T; ++t) {
    const unsigned short* u_cur = u_ex + (size_t)(t & 1) * (RNN_M * RNN_N);
    unsigned short*       u_nxt = u_ex + (size_t)((t + 1) & 1) * (RNN_M * RNN_N);

    // Stage I[:, :, t] (128 scattered floats, stride T) into LDS with the
    // CDNA5 async global->LDS path, tracked by ASYNCcnt.
    if (tid < RNN_M * 2) {
      const float* gp = I + (size_t)tid * (size_t)T + (size_t)t;
      unsigned lds = (unsigned)(size_t)&Ish[tid];   // low 32 bits of flat addr = LDS offset
      asm volatile("global_load_async_to_lds_b32 %0, %1, off"
                   :: "v"(lds), "v"(gp) : "memory");
      asm volatile("s_wait_asynccnt 0x0" ::: "memory");
    }

    // Recurrent matmul: two 16x16 f32 tiles (M-tiles 2*mpair, 2*mpair+1).
    // A row = mtile*16 + l16; per lane chunks at K = kt*32 + (hi?8:0) and +16.
    const unsigned short* a0p = u_cur + (size_t)((2 * mpair) * 16 + l16) * RNN_N + hi * 8;
    const unsigned short* a1p = a0p + (size_t)16 * RNN_N;

    v8f acc0 = {}, acc1 = {};
    v16bf an0 = load_a(a0p), an1 = load_a(a1p);        // prefetch kt = 0
    #pragma unroll
    for (int kt = 0; kt < 16; ++kt) {
      const v16bf c0 = an0, c1 = an1;
      if (kt < 15) {                                   // double-buffer next A pair
        an0 = load_a(a0p + (kt + 1) * 32);
        an1 = load_a(a1p + (kt + 1) * 32);
      }
      acc0 = __builtin_amdgcn_wmma_f32_16x16x32_bf16(false, c0, false, Breg[kt],
                                                     (short)0, acc0, false, false);
      acc1 = __builtin_amdgcn_wmma_f32_16x16x32_bf16(false, c1, false, Breg[kt],
                                                     (short)0, acc1, false, false);
    }

    __syncthreads();            // Ish visible to all waves

    // Euler update + tanh; emit x_{t+1}, u_{t+1} at index t+1; publish bf16 u.
    const bool last = (t == T - 1);
    #pragma unroll
    for (int a = 0; a < 2; ++a) {
      #pragma unroll
      for (int j = 0; j < 8; ++j) {
        const int m     = (2 * mpair + a) * 16 + hi * 8 + j;
        const float rec = a ? acc1[j] : acc0[j];
        const float inp = Ish[2 * m + 0] * win0 + Ish[2 * m + 1] * win1;
        const float xn  = xr[a][j] + ALPHA * (rec + inp + bbn - xr[a][j]);
        const float un  = tanhf(xn);
        xr[a][j] = xn;
        u_nxt[(size_t)m * RNN_N + ngl] = (unsigned short)bf16rne(un);
        if (!last) {
          const size_t o = obase[a] + (size_t)j * NT + (size_t)(t + 1);
          x_out[o] = xn;
          u_out[o] = un;
        } else {
          u_fin[(size_t)m * RNN_N + ngl] = un;   // u(T) for the t = T-1 readout
        }
      }
    }

    grid_barrier(bar_cnt, bar_epoch, (unsigned)(t + 2));
  }
}

// Pass 2: y[m, :, t] = u_{t+1}[m, :] @ W_out^T ; u_{t+1} = stored u[:, :, t+1] (or u_fin).
// Threads map to consecutive t -> coalesced reads along the T axis.
__global__ void ctrnn_readout(const float* __restrict__ u_out, const float* __restrict__ u_fin,
                              const float* __restrict__ W_out, float* __restrict__ y, int T)
{
  const int t = blockIdx.x * blockDim.x + threadIdx.x;
  const int m = blockIdx.y;
  if (t >= T) return;
  const float* p; size_t stride;
  if (t < T - 1) { p = u_out + (size_t)m * RNN_N * (size_t)T + (size_t)(t + 1); stride = (size_t)T; }
  else           { p = u_fin + (size_t)m * RNN_N;                               stride = 1; }
  float a0 = 0.f, a1 = 0.f;
  for (int n = 0; n < RNN_N; ++n) {
    const float uv = p[(size_t)n * stride];
    a0 += uv * W_out[n];            // W_out[0][n]
    a1 += uv * W_out[RNN_N + n];    // W_out[1][n]
  }
  y[((size_t)m * 2 + 0) * (size_t)T + (size_t)t] = a0;
  y[((size_t)m * 2 + 1) * (size_t)T + (size_t)t] = a1;
}

extern "C" void kernel_launch(void* const* d_in, const int* in_sizes, int n_in,
                              void* d_out, int out_size, void* d_ws, size_t ws_size,
                              hipStream_t stream)
{
  const float* I     = (const float*)d_in[0];
  const float* x0    = (const float*)d_in[1];
  const float* W_in  = (const float*)d_in[2];
  const float* W_rec = (const float*)d_in[3];
  const float* b     = (const float*)d_in[4];
  const float* W_out = (const float*)d_in[5];
  const int T = in_sizes[0] / (RNN_M * 2);    // I is (64, 2, T)

  float* x_out = (float*)d_out;
  float* u_out = x_out + (size_t)RNN_M * RNN_N * (size_t)T;
  float* y_out = u_out + (size_t)RNN_M * RNN_N * (size_t)T;

  char* ws = (char*)d_ws;
  unsigned*       bar_cnt   = (unsigned*)ws;
  unsigned*       bar_epoch = (unsigned*)(ws + 128);
  unsigned short* u_ex      = (unsigned short*)(ws + 256);                       // 2 x 64KB bf16
  float*          u_fin     = (float*)(ws + 256 + 2 * (size_t)RNN_M * RNN_N * sizeof(unsigned short));

  hipMemsetAsync(d_ws, 0, 256, stream);       // reset barrier state every call (capture-safe)

  ctrnn_persistent<<<NWG, 256, 0, stream>>>(I, x0, W_in, W_rec, b,
                                            x_out, u_out, u_ex, u_fin,
                                            bar_cnt, bar_epoch, T);

  dim3 grid((T + 255) / 256, RNN_M);
  ctrnn_readout<<<grid, 256, 0, stream>>>(u_out, u_fin, W_out, y_out, T);
}
